// TuneInHGT_54571854463736
// MI455X (gfx1250) — matrix operand order repaired and checked by reading, not compiled, not run on hardware
//
#include <hip/hip_runtime.h>
#include <hip/hip_bf16.h>
#include <math.h>

// ---------------------------------------------------------------------------
// HGT forward for MI455X (gfx1250, wave32).
// Dense GEMMs -> v_wmma_f32_16x16x32_f16 (f32 accum).
//   A tiles: Tensor Data Mover (tensor_load_to_lds) DMA, f32 in LDS, padded
//            rows (144B stride) for conflict-free ds_load_b128; f32->f16
//            conversion happens in registers during fragment build.
//   B tiles: store-transposed f16 in LDS so fragments are contiguous b128.
// Edge phase -> wave-per-edge gathers + f32 atomics (L2-resident).
// ---------------------------------------------------------------------------

typedef __attribute__((ext_vector_type(16))) _Float16 v16h;
typedef __attribute__((ext_vector_type(8)))  _Float16 v8h;
typedef __attribute__((ext_vector_type(8)))  float    v8f;
typedef __attribute__((ext_vector_type(4)))  float    f4;
typedef __attribute__((ext_vector_type(4)))  unsigned uint4v;
typedef __attribute__((ext_vector_type(8)))  int      int8v;
typedef __attribute__((ext_vector_type(4)))  int      int4v;

#define BM 128
#define BN 64
#define BK 32
#define LDA_F 36   // f32 LDS row stride: 144B = 9 x 16B (set by TDM pad: 32+4 DW)
#define LDB_S 40   // f16 LDS row stride: 80B = 5 x 16B

// Generic tiled WMMA GEMM: C[z] = act(A[z] * W[z] + bias), f32 in/out, f16 MACs.
// gridDim.z batches heads for the 32x32 relation transforms.
__global__ __launch_bounds__(256)
void gemm_wmma(const float* __restrict__ A, int lda, long long aZ,
               const float* __restrict__ W, int ldw, long long wZ,
               const float* __restrict__ bias,
               float* __restrict__ C, int ldc, long long cZ,
               int M, int K, int N, int act)
{
    // A tile (f32, natural order) filled by the Tensor Data Mover.
    __shared__ __attribute__((aligned(16))) float    Asf[BM][LDA_F];
    // B tile stored TRANSPOSED f16: BsT[n][k], lane fragments are contiguous.
    __shared__ __attribute__((aligned(16))) _Float16 BsT[BN][LDB_S];

    A += (long long)blockIdx.z * aZ;
    W += (long long)blockIdx.z * wZ;
    C += (long long)blockIdx.z * cZ;

    const int tid  = threadIdx.x;
    const int wave = tid >> 5;
    const int lane = tid & 31;
    const int m0   = blockIdx.x * BM;
    const int n0   = blockIdx.y * BN;

    const unsigned ldsA = (unsigned)(unsigned long long)(const void*)&Asf[0][0];

    v8f acc[4] = {};

    for (int k0 = 0; k0 < K; k0 += BK) {
        // ---- A tile via TDM: one tensor_load_to_lds per K-step (wave 0) ----
        if (tid < 32) {
            unsigned long long ga =
                (unsigned long long)(const void*)(A + (long long)m0 * lda + k0);
            unsigned td0 = (unsigned)(K - k0);   // line length for OOB check
            unsigned td1 = (unsigned)(M - m0);   // rows left: OOB rows -> 0

            uint4v g0;
            g0[0] = 1u;                                    // count=1 (user D#)
            g0[1] = ldsA;                                  // lds_addr (bytes)
            g0[2] = (unsigned)ga;                          // global_addr[31:0]
            g0[3] = (unsigned)((ga >> 32) & 0x01FFFFFFu)   // global_addr[56:32]
                  | (2u << 30);                            // type = 2 (image)

            int8v g1;
            g1[0] = (int)((2u << 16)     // data_size = 4B
                        | (1u << 20)     // pad_enable
                        | (4u << 22)     // pad_interval: every 32 DWORDs
                        | (3u << 25));   // pad_amount: 4 DWORDs -> stride 36
            g1[1] = (int)((td0 & 0xFFFFu) << 16);              // tdim0[15:0]
            g1[2] = (int)((td0 >> 16) | ((td1 & 0xFFFFu) << 16)); // tdim0 hi|tdim1 lo
            g1[3] = (int)((td1 >> 16) | (32u << 16));          // tdim1 hi | tile_dim0=32
            g1[4] = (int)128u;                                 // tile_dim1=128
            g1[5] = (int)(unsigned)lda;                        // tensor_dim0_stride
            g1[6] = 0;
            g1[7] = 0;
            int4v gz = {0, 0, 0, 0};
#if __clang_major__ >= 23
            int8v gz8 = {0, 0, 0, 0, 0, 0, 0, 0};
            __builtin_amdgcn_tensor_load_to_lds(g0, g1, gz, gz, gz8, 0);
#else
            __builtin_amdgcn_tensor_load_to_lds(g0, g1, gz, gz, 0);
#endif
        }

        // ---- stage B transposed: 32x64 f32 -> BsT[n][k] f16 -----------------
#pragma unroll
        for (int j = 0; j < 2; ++j) {
            int i  = tid + j * 256;          // vec4 index, 512 total
            int r  = i >> 4;                 // k within tile
            int c4 = (i & 15) * 4;           // n within tile
            f4 v = {0.f, 0.f, 0.f, 0.f};
            if (n0 + c4 < N) {               // N % 4 == 0 -> all-or-nothing
                v = *(const f4*)&W[(long long)(k0 + r) * ldw + n0 + c4];
                if (k0 + BK < K)             // global_prefetch_b8
                    __builtin_prefetch(&W[(long long)(k0 + BK + r) * ldw + n0 + c4], 0, 0);
            }
            BsT[c4 + 0][r] = (_Float16)v.x;
            BsT[c4 + 1][r] = (_Float16)v.y;
            BsT[c4 + 2][r] = (_Float16)v.z;
            BsT[c4 + 3][r] = (_Float16)v.w;
        }

        __builtin_amdgcn_s_wait_tensorcnt(0);   // TDM data resident in LDS
        __syncthreads();

        // ---- A fragment: 4x ds_load_b128 (f32) + in-register cvt to f16 -----
        // lanes 0-15: M=lane, kbase=0 ; lanes 16-31: M=lane-16, kbase=8.
        const int mrow  = wave * 16 + (lane & 15);
        const int kbase = (lane >> 4) * 8;
        f4 a0 = *(const f4*)&Asf[mrow][kbase];
        f4 a1 = *(const f4*)&Asf[mrow][kbase + 4];
        f4 a2 = *(const f4*)&Asf[mrow][kbase + 16];
        f4 a3 = *(const f4*)&Asf[mrow][kbase + 20];
        v16h a;
#pragma unroll
        for (int j = 0; j < 4; ++j) {
            a[j]      = (_Float16)a0[j];
            a[j + 4]  = (_Float16)a1[j];
            a[j + 8]  = (_Float16)a2[j];
            a[j + 12] = (_Float16)a3[j];
        }

        // ---- B fragments: two 8-half k runs (kb2, kb2+16) -> 2x ds_load_b128 per tile --
        const int nc  = lane & 15;
        const int kb2 = (lane >> 4) * 8;
#pragma unroll
        for (int t = 0; t < 4; ++t) {
            v8h blo = *(const v8h*)&BsT[t * 16 + nc][kb2];
            v8h bhi = *(const v8h*)&BsT[t * 16 + nc][kb2 + 16];
            v16h b = __builtin_shufflevector(blo, bhi,
                     0,1,2,3,4,5,6,7, 8,9,10,11,12,13,14,15);
            acc[t] = __builtin_amdgcn_wmma_f32_16x16x32_f16(
                false, a, false, b, (short)0, acc[t], false, false);
        }
        __syncthreads();
    }

    // Epilogue. D layout: col = lane&15 ; row = r + (lane>=16 ? 8 : 0).
    const int col   = lane & 15;
    const int rhalf = (lane >> 4) * 8;
#pragma unroll
    for (int t = 0; t < 4; ++t) {
        int gn = n0 + t * 16 + col;
        if (gn >= N) continue;
        float bv = bias ? bias[gn] : 0.0f;
#pragma unroll
        for (int r = 0; r < 8; ++r) {
            int gm = m0 + wave * 16 + rhalf + r;
            if (gm < M) {
                float v = acc[t][r] + bv;
                if (act == 1) v = fmaxf(v, 0.0f);
                C[(long long)gm * ldc + gn] = v;
            }
        }
    }
}

// ---------------------------------------------------------------------------
__device__ inline void atomicMaxFloat(float* addr, float v) {
    unsigned* u = (unsigned*)addr;
    unsigned old = __float_as_uint(*addr);
    while (__uint_as_float(old) < v) {
        unsigned assumed = old;
        old = atomicCAS(u, assumed, __float_as_uint(v));
        if (old == assumed) break;
    }
}

__device__ inline float waveSum(float p) {
#pragma unroll
    for (int off = 16; off > 0; off >>= 1) p += __shfl_xor(p, off, 32);
    return p;
}

// score[e,h] = (q_d[dst] . krel[src])_h * prel[h] * SCALE ; segmax via atomics.
__global__ __launch_bounds__(256)
void edge_score(const int* __restrict__ ei, int E,
                const float* __restrict__ q_d,
                const float* __restrict__ krel,
                const float* __restrict__ prel,
                float* __restrict__ score,
                float* __restrict__ segmax)
{
    int e = blockIdx.x * (blockDim.x >> 5) + (threadIdx.x >> 5);
    if (e >= E) return;
    int lane = threadIdx.x & 31;
    int s = ei[e];
    int d = ei[E + e];
    const float* qrow = q_d  + (long long)d * 128;
    const float* krow = krel + (long long)s * 128;
#pragma unroll
    for (int h = 0; h < 4; ++h) {
        float p  = qrow[h * 32 + lane] * krow[h * 32 + lane];
        float sc = waveSum(p) * prel[h] * 0.17677669529663687f; // 1/sqrt(32)
        if (lane == 0) {
            score[(long long)e * 4 + h] = sc;
            atomicMaxFloat(&segmax[(long long)d * 4 + h], sc);
        }
    }
}

// alpha = exp(score - segmax[dst]) ; segsum += alpha  (thread per (e,h))
__global__ __launch_bounds__(256)
void edge_expsum(const int* __restrict__ ei, int E,
                 float* __restrict__ score,
                 const float* __restrict__ segmax,
                 float* __restrict__ segsum)
{
    long long i = (long long)blockIdx.x * blockDim.x + threadIdx.x;
    if (i >= (long long)E * 4) return;
    int e = (int)(i >> 2), h = (int)(i & 3);
    int d = ei[E + e];
    float a = expf(score[i] - segmax[(long long)d * 4 + h]);
    score[i] = a;
    atomicAdd(&segsum[(long long)d * 4 + h], a);
}

// out[dst] += vrel[src] * alpha / (segsum[dst] + eps)   (wave per edge)
__global__ __launch_bounds__(256)
void edge_message(const int* __restrict__ ei, int E,
                  const float* __restrict__ alpha,
                  const float* __restrict__ segsum,
                  const float* __restrict__ vrel,
                  float* __restrict__ outacc)
{
    int e = blockIdx.x * (blockDim.x >> 5) + (threadIdx.x >> 5);
    if (e >= E) return;
    int lane = threadIdx.x & 31;
    int s = ei[e];
    int d = ei[E + e];
#pragma unroll
    for (int h = 0; h < 4; ++h) {
        float w = alpha[(long long)e * 4 + h] /
                  (segsum[(long long)d * 4 + h] + 1e-16f);
        float m = vrel[(long long)s * 128 + h * 32 + lane] * w;
        atomicAdd(&outacc[(long long)d * 128 + h * 32 + lane], m);
    }
}

// ---------------------------------------------------------------------------
__global__ void fill_kernel(float* __restrict__ p, long long n, float v) {
    long long i = (long long)blockIdx.x * blockDim.x + threadIdx.x;
    if (i < n) p[i] = v;
}

// x = gelu(x * scale)  -- exact gelu, scale implements cross-relation mean
__global__ void gelu_scale(float* __restrict__ p, long long n, float scale) {
    long long i = (long long)blockIdx.x * blockDim.x + threadIdx.x;
    if (i < n) {
        float x = p[i] * scale;
        p[i] = 0.5f * x * (1.0f + erff(x * 0.70710678118654752f));
    }
}

// h = a*o + (1-a)*h, a = sigmoid(skip)
__global__ void skip_blend(const float* __restrict__ o,
                           const float* __restrict__ hin,
                           const float* __restrict__ skipP,
                           float* __restrict__ hout, long long n) {
    long long i = (long long)blockIdx.x * blockDim.x + threadIdx.x;
    if (i < n) {
        float a = 1.0f / (1.0f + expf(-skipP[0]));
        hout[i] = a * o[i] + (1.0f - a) * hin[i];
    }
}

// LayerNorm over 128 features, wave per row (4 elems/lane)
__global__ __launch_bounds__(256)
void layer_norm(float* __restrict__ h, int rows) {
    int row = blockIdx.x * (blockDim.x >> 5) + (threadIdx.x >> 5);
    if (row >= rows) return;
    int lane = threadIdx.x & 31;
    float x[4], s = 0.0f;
#pragma unroll
    for (int j = 0; j < 4; ++j) {
        x[j] = h[(long long)row * 128 + j * 32 + lane];
        s += x[j];
    }
    float mu = waveSum(s) * (1.0f / 128.0f);
    float v = 0.0f;
#pragma unroll
    for (int j = 0; j < 4; ++j) { float c = x[j] - mu; v += c * c; }
    float inv = rsqrtf(waveSum(v) * (1.0f / 128.0f) + 1e-5f);
#pragma unroll
    for (int j = 0; j < 4; ++j)
        h[(long long)row * 128 + j * 32 + lane] = (x[j] - mu) * inv;
}

// out[row] = dot(A[row,:K], w) + b   (wave per row; K <= 128)
__global__ __launch_bounds__(256)
void rowdot(const float* __restrict__ A, int lda, int K,
            const float* __restrict__ w, const float* __restrict__ b,
            float* __restrict__ out, int rows) {
    int row = blockIdx.x * (blockDim.x >> 5) + (threadIdx.x >> 5);
    if (row >= rows) return;
    int lane = threadIdx.x & 31;
    float s = 0.0f;
    for (int k = lane; k < K; k += 32)
        s += A[(long long)row * lda + k] * w[k];
    s = waveSum(s);
    if (lane == 0) out[row] = s + b[0];
}

// ---------------------------------------------------------------------------
static inline void launch_gemm(const float* A, int lda, long long aZ,
                               const float* W, int ldw, long long wZ,
                               const float* bias,
                               float* C, int ldc, long long cZ,
                               int M, int K, int N, int act, int nz,
                               hipStream_t s) {
    dim3 g((M + BM - 1) / BM, (N + BN - 1) / BN, nz);
    gemm_wmma<<<g, 256, 0, s>>>(A, lda, aZ, W, ldw, wZ, bias, C, ldc, cZ,
                                M, K, N, act);
}

extern "C" void kernel_launch(void* const* d_in, const int* in_sizes, int n_in,
                              void* d_out, int out_size, void* d_ws, size_t ws_size,
                              hipStream_t stream) {
    (void)in_sizes; (void)n_in; (void)out_size; (void)ws_size;

    // ---- problem constants -------------------------------------------------
    const int NT[3]   = {50000, 100000, 2000};     // user, post, club
    const int NO[3]   = {0, 50000, 150000};        // node offsets in fused buf
    const int TN      = 152000;
    const int FEAT_[3]= {256, 512, 128};
    const int EN[5]   = {400000, 300000, 100000, 400000, 100000};
    const int ST[5]   = {0, 0, 0, 1, 2};           // source node type per rel
    const int DT[5]   = {1, 0, 2, 0, 0};           // dest node type per rel
    const int RO[5]   = {0, 50000, 100000, 150000, 250000};  // src slot offset
    const int DO_[5]  = {0, 100000, 150000, 152000, 202000}; // dst slot offset
    const long long EO[5] = {0, 400000, 700000, 800000, 1200000};
    const int SR = 252000, DR = 252000;
    const long long ETOT = 1300000;
    const float MEAN_SCALE[3] = {1.0f / 3.0f, 1.0f, 1.0f}; // relations per dst

    // ---- inputs ------------------------------------------------------------
    const float* X[3] = {(const float*)d_in[0], (const float*)d_in[1],
                         (const float*)d_in[2]};
    const int* EI[5]  = {(const int*)d_in[3], (const int*)d_in[4],
                         (const int*)d_in[5], (const int*)d_in[6],
                         (const int*)d_in[7]};

    int pi = 8;
    auto F = [&](int i) { return (const float*)d_in[i]; };
    const float *inW[3], *inB[3];
    for (int t = 0; t < 3; ++t) { inW[t] = F(pi++); inB[t] = F(pi++); }
    const float *kW[2][3], *kb[2][3], *qW[2][3], *qb[2][3],
                *vW[2][3], *vb[2][3], *aW[2][3], *ab[2][3], *skp[2][3];
    const float *arel[2][5], *mrel[2][5], *prel[2][5];
    for (int l = 0; l < 2; ++l) {
        for (int t = 0; t < 3; ++t) {
            kW[l][t] = F(pi++); kb[l][t] = F(pi++);
            qW[l][t] = F(pi++); qb[l][t] = F(pi++);
            vW[l][t] = F(pi++); vb[l][t] = F(pi++);
            aW[l][t] = F(pi++); ab[l][t] = F(pi++);
            skp[l][t] = F(pi++);
        }
        for (int r = 0; r < 5; ++r) {
            arel[l][r] = F(pi++); mrel[l][r] = F(pi++); prel[l][r] = F(pi++);
        }
    }
    const float* embW = F(pi++); const float* embB = F(pi++);
    const float* s1W  = F(pi++); const float* s1B  = F(pi++);
    const float* s2W  = F(pi++); const float* s2B  = F(pi++);
    const float* psW  = F(pi++); const float* psB  = F(pi++);

    // ---- workspace layout (fp32) ------------------------------------------
    float* ws     = (float*)d_ws;
    float* h      = ws;                              // TN*128
    float* htmp   = h      + (long long)TN * 128;    // TN*128
    float* kbuf   = htmp   + (long long)TN * 128;    // TN*128
    float* qbuf   = kbuf   + (long long)TN * 128;    // TN*128
    float* vbuf   = qbuf   + (long long)TN * 128;    // TN*128
    float* krel   = vbuf   + (long long)TN * 128;    // SR*128
    float* vrel   = krel   + (long long)SR * 128;    // SR*128
    float* outacc = vrel   + (long long)SR * 128;    // TN*128
    float* alpha  = outacc + (long long)TN * 128;    // ETOT*4
    float* segmax = alpha  + ETOT * 4;               // DR*4
    float* segsum = segmax + (long long)DR * 4;      // DR*4

    auto fill = [&](float* p, long long n, float v) {
        fill_kernel<<<dim3((unsigned)((n + 255) / 256)), 256, 0, stream>>>(p, n, v);
    };

    // ---- input projections: h = relu(x @ inW + inB) -----------------------
    for (int t = 0; t < 3; ++t)
        launch_gemm(X[t], FEAT_[t], 0, inW[t], 128, 0, inB[t],
                    h + (long long)NO[t] * 128, 128, 0,
                    NT[t], FEAT_[t], 128, /*relu*/1, 1, stream);

    // ---- HGT layers --------------------------------------------------------
    for (int l = 0; l < 2; ++l) {
        // K/Q/V projections per node type
        for (int t = 0; t < 3; ++t) {
            const float* hi = h + (long long)NO[t] * 128;
            launch_gemm(hi, 128, 0, kW[l][t], 128, 0, kb[l][t],
                        kbuf + (long long)NO[t] * 128, 128, 0,
                        NT[t], 128, 128, 0, 1, stream);
            launch_gemm(hi, 128, 0, qW[l][t], 128, 0, qb[l][t],
                        qbuf + (long long)NO[t] * 128, 128, 0,
                        NT[t], 128, 128, 0, 1, stream);
            launch_gemm(hi, 128, 0, vW[l][t], 128, 0, vb[l][t],
                        vbuf + (long long)NO[t] * 128, 128, 0,
                        NT[t], 128, 128, 0, 1, stream);
        }
        // Per-relation, per-head 32x32 transforms (gridDim.z = 4 heads)
        for (int r = 0; r < 5; ++r) {
            int st = ST[r];
            launch_gemm(kbuf + (long long)NO[st] * 128, 128, /*aZ=*/32,
                        arel[l][r], 32, /*wZ=*/1024, nullptr,
                        krel + (long long)RO[r] * 128, 128, /*cZ=*/32,
                        NT[st], 32, 32, 0, /*nz=*/4, stream);
            launch_gemm(vbuf + (long long)NO[st] * 128, 128, 32,
                        mrel[l][r], 32, 1024, nullptr,
                        vrel + (long long)RO[r] * 128, 128, 32,
                        NT[st], 32, 32, 0, 4, stream);
        }
        // Edge phase
        fill(outacc, (long long)TN * 128, 0.0f);
        fill(segmax, (long long)DR * 4, -1e30f);
        fill(segsum, (long long)DR * 4, 0.0f);
        for (int r = 0; r < 5; ++r) {
            int dt = DT[r];
            int E = EN[r];
            dim3 gw((E + 7) / 8);
            edge_score<<<gw, 256, 0, stream>>>(
                EI[r], E, qbuf + (long long)NO[dt] * 128,
                krel + (long long)RO[r] * 128, prel[l][r],
                alpha + EO[r] * 4, segmax + (long long)DO_[r] * 4);
            long long eh = (long long)E * 4;
            edge_expsum<<<dim3((unsigned)((eh + 255) / 256)), 256, 0, stream>>>(
                EI[r], E, alpha + EO[r] * 4,
                segmax + (long long)DO_[r] * 4, segsum + (long long)DO_[r] * 4);
            edge_message<<<gw, 256, 0, stream>>>(
                EI[r], E, alpha + EO[r] * 4, segsum + (long long)DO_[r] * 4,
                vrel + (long long)RO[r] * 128,
                outacc + (long long)NO[dt] * 128);
        }
        // Per node type: mean over relations + GELU, A-projection, skip, LN
        for (int t = 0; t < 3; ++t) {
            long long n = (long long)NT[t] * 128;
            float* oa = outacc + (long long)NO[t] * 128;
            gelu_scale<<<dim3((unsigned)((n + 255) / 256)), 256, 0, stream>>>(
                oa, n, MEAN_SCALE[t]);
            launch_gemm(oa, 128, 0, aW[l][t], 128, 0, ab[l][t],
                        htmp + (long long)NO[t] * 128, 128, 0,
                        NT[t], 128, 128, 0, 1, stream);
            skip_blend<<<dim3((unsigned)((n + 255) / 256)), 256, 0, stream>>>(
                htmp + (long long)NO[t] * 128, h + (long long)NO[t] * 128,
                skp[l][t], h + (long long)NO[t] * 128, n);
            layer_norm<<<dim3((NT[t] + 7) / 8), 256, 0, stream>>>(
                h + (long long)NO[t] * 128, NT[t]);
        }
    }

    // ---- output heads ------------------------------------------------------
    float* out = (float*)d_out;
    // user embeddings: [50000, 64]
    launch_gemm(h, 128, 0, embW, 64, 0, embB, out, 64, 0,
                NT[0], 128, 64, 0, 1, stream);
    // user scores: relu(h_user @ s1W + b) @ s2W + b   (reuse kbuf as 50000x32)
    launch_gemm(h, 128, 0, s1W, 32, 0, s1B, kbuf, 32, 0,
                NT[0], 128, 32, /*relu*/1, 1, stream);
    rowdot<<<dim3((NT[0] + 7) / 8), 256, 0, stream>>>(
        kbuf, 32, 32, s2W, s2B, out + (long long)NT[0] * 64, NT[0]);
    // post scores: h_post @ psW + b
    rowdot<<<dim3((NT[1] + 7) / 8), 256, 0, stream>>>(
        h + (long long)NO[1] * 128, 128, 128, psW, psB,
        out + (long long)NT[0] * 64 + NT[0], NT[1]);
}
